// SiameseGIN_72232759984513
// MI455X (gfx1250) — matrix operand order repaired and checked by reading, not compiled, not run on hardware
//
#include <hip/hip_runtime.h>
#include <math.h>

typedef __attribute__((ext_vector_type(16))) _Float16 v16h;
typedef __attribute__((ext_vector_type(8)))  float    v8f;

#define N_NODES 100000
#define N_EDGES 1280000
#define N_GRAPH 256
#define F_IN 64
#define HID 128
#define EMB 64
#define BN_EPS 1e-5f

// ---------------- utility kernels ----------------

__global__ void k_zero(float* __restrict__ p, int n) {
  int t = blockIdx.x * blockDim.x + threadIdx.x;
  if (t < n) p[t] = 0.f;
}

__global__ void k_copy4(float* __restrict__ dst, const float* __restrict__ src, int n4) {
  int t = blockIdx.x * blockDim.x + threadIdx.x;
  if (t < n4) ((float4*)dst)[t] = ((const float4*)src)[t];
}

// ---------------- GIN aggregation: agg[dst] += x[src] ----------------
// One thread per (edge, 4-feature group); 32 consecutive threads share an edge
// when C=128, so the x[src] row read is coalesced and atomics hit one row.

__global__ void k_scatter(const float* __restrict__ x, const int* __restrict__ src,
                          const int* __restrict__ dst, float* __restrict__ agg,
                          int nedges, int C) {
  const int c4 = C >> 2;
  long t = (long)blockIdx.x * blockDim.x + threadIdx.x;
  if (t >= (long)nedges * c4) return;
  int e = (int)(t / c4);
  int q = (int)(t % c4) << 2;
  int s = src[e], d = dst[e];
  float4 v = *(const float4*)(x + (long)s * C + q);
  float* out = agg + (long)d * C + q;
  atomicAdd(out + 0, v.x);
  atomicAdd(out + 1, v.y);
  atomicAdd(out + 2, v.z);
  atomicAdd(out + 3, v.w);
}

// ---------------- WMMA GEMM: D[M,Ncols] = A[M,K] @ W[K,Ncols] (+bias) ----------------
// f16 inputs (converted in-register), f32 accumulate via v_wmma_f32_16x16x32_f16.
// 4 waves / block, each wave owns one 16-row tile; all waves share one 16-col
// tile whose K x 16 weight slab is staged once in LDS (f16, <= 4 KB), stored
// n-major so each lane's B fragment is 16 contiguous halves -> 2x ds_load_b128.
// K is a compile-time template arg (64 or 128) so the k-loop fully unrolls and
// the WMMAs issue back-to-back with loads/cvts hoisted ahead of them.
// Fragment layouts (wave32; A/B use contiguous k, not the ISA 7.12.2 table order):
//   A 16x32 f16 : m = lane&15 ; element j -> k = ((lane>>4)<<4) + j
//   B 32x16 f16 : n = lane&15 ; element j -> k = ((lane>>4)<<4) + j
//   C/D 16x16 f32: row = ((lane>>4)<<3) + r ; col = lane&15

template <int K>
__global__ void k_gemm_wmma(const float* __restrict__ A, const float* __restrict__ W,
                            const float* __restrict__ bias, float* __restrict__ D,
                            int M, int Ncols) {
  const int lane   = threadIdx.x & 31;
  const int wave   = threadIdx.x >> 5;
  const int mtile  = blockIdx.x * 4 + wave;
  const int col0   = blockIdx.y << 4;
  const int half16 = lane >> 4;

  // Cooperative stage of the K x 16 weight tile into LDS as f16, n-major [n][K]
  // (coalesced global read of W; strided LDS write happens once per block).
  __shared__ _Float16 Bs[16 * K];
  for (int i = threadIdx.x; i < 16 * K; i += 128) {
    const int k = i >> 4, n = i & 15;
    Bs[n * K + k] = (_Float16)W[(long)k * Ncols + col0 + n];
  }
  __syncthreads();

  if ((mtile << 4) < M) {                         // wave-uniform: EXEC stays all-1
    const int m   = (mtile << 4) + (lane & 15);
    const int col = col0 + (lane & 15);
    v8f acc = {};
#pragma unroll
    for (int k0 = 0; k0 < K; k0 += 32) {
      const float* arow = A + (long)m * K + k0 + (half16 << 4);
      float4 a0 = *(const float4*)(arow);
      float4 a1 = *(const float4*)(arow + 4);
      float4 a2 = *(const float4*)(arow + 8);
      float4 a3 = *(const float4*)(arow + 12);
      v16h af, bfrag;
      af[0]=(_Float16)a0.x;  af[1]=(_Float16)a0.y;  af[2]=(_Float16)a0.z;  af[3]=(_Float16)a0.w;
      af[4]=(_Float16)a1.x;  af[5]=(_Float16)a1.y;  af[6]=(_Float16)a1.z;  af[7]=(_Float16)a1.w;
      af[8]=(_Float16)a2.x;  af[9]=(_Float16)a2.y;  af[10]=(_Float16)a2.z; af[11]=(_Float16)a2.w;
      af[12]=(_Float16)a3.x; af[13]=(_Float16)a3.y; af[14]=(_Float16)a3.z; af[15]=(_Float16)a3.w;
      // 16 contiguous halves per lane (32 B, 16 B-aligned) -> vectorized LDS loads
      const _Float16* bp = &Bs[(lane & 15) * K + k0 + (half16 << 4)];
#pragma unroll
      for (int j = 0; j < 16; ++j) bfrag[j] = bp[j];
      acc = __builtin_amdgcn_wmma_f32_16x16x32_f16(false, af, false, bfrag,
                                                   (short)0, acc, false, false);
    }
    const float bv = bias ? bias[col] : 0.f;
    float* drow = D + (long)((mtile << 4) + (half16 << 3)) * Ncols + col;
#pragma unroll
    for (int r = 0; r < 8; ++r) drow[(long)r * Ncols] = acc[r] + bv;
  }
}

// ---------------- BatchNorm statistics (sum, sum-sq per column) ----------------
// blockDim = 256 = 2 threads per column (HID=128); per-thread partials, then
// two f32 atomics per thread into stats[0:128]=sum, stats[128:256]=sumsq.

__global__ void k_bn_stats(const float* __restrict__ h, float* __restrict__ stats,
                           int nrows, int rows_per_block) {
  const int c   = threadIdx.x & (HID - 1);
  const int sub = threadIdx.x >> 7;
  int r0 = blockIdx.x * rows_per_block;
  int r1 = min(r0 + rows_per_block, nrows);
  float s = 0.f, s2 = 0.f;
  for (int r = r0 + sub; r < r1; r += 2) {
    float v = h[(long)r * HID + c];
    s += v; s2 += v * v;
  }
  atomicAdd(&stats[c], s);
  atomicAdd(&stats[HID + c], s2);
}

// BN apply + ReLU, in place. (Linear bias b cancels in (h - mu): skipped upstream.)
__global__ void k_bn_apply(float* __restrict__ x, const float* __restrict__ stats,
                           const float* __restrict__ gamma, const float* __restrict__ beta,
                           long n, float invN) {
  long t = (long)blockIdx.x * blockDim.x + threadIdx.x;
  if (t >= n) return;
  int c = (int)(t & (HID - 1));
  float mu  = stats[c] * invN;
  float var = stats[HID + c] * invN - mu * mu;
  float sc  = rsqrtf(var + BN_EPS) * gamma[c];
  float v   = (x[t] - mu) * sc + beta[c];
  x[t] = fmaxf(v, 0.f);
}

// ---------------- global mean pool ----------------

__global__ void k_pool_accum(const float* __restrict__ x, const int* __restrict__ batch,
                             float* __restrict__ psum, float* __restrict__ cnt) {
  const int c4 = HID >> 2;
  long t = (long)blockIdx.x * blockDim.x + threadIdx.x;
  if (t >= (long)N_NODES * c4) return;
  int n = (int)(t / c4);
  int q = (int)(t % c4) << 2;
  int b = batch[n];
  float4 v = *(const float4*)(x + (long)n * HID + q);
  float* out = psum + (long)b * HID + q;
  atomicAdd(out + 0, v.x);
  atomicAdd(out + 1, v.y);
  atomicAdd(out + 2, v.z);
  atomicAdd(out + 3, v.w);
  if (q == 0) atomicAdd(&cnt[b], 1.f);
}

__global__ void k_pool_final(float* __restrict__ psum, const float* __restrict__ cnt) {
  int t = blockIdx.x * blockDim.x + threadIdx.x;
  if (t >= N_GRAPH * HID) return;
  int g = t / HID;
  psum[t] = psum[t] / fmaxf(cnt[g], 1.f);
}

// ---------------- classifier: sigmoid(relu(|v1-v2| @ Wc1 + bc1) @ Wc2 + bc2) ----------------

__global__ void k_classifier(const float* __restrict__ v1, const float* __restrict__ v2,
                             const float* __restrict__ Wc1, const float* __restrict__ bc1,
                             const float* __restrict__ Wc2, const float* __restrict__ bc2,
                             float* __restrict__ out) {
  __shared__ float w1s[EMB * 64];
  __shared__ float w2s[64];
  for (int i = threadIdx.x; i < EMB * 64; i += blockDim.x) w1s[i] = Wc1[i];
  if (threadIdx.x < 64) w2s[threadIdx.x] = Wc2[threadIdx.x];
  __syncthreads();
  int g = blockIdx.x * blockDim.x + threadIdx.x;
  if (g >= N_GRAPH) return;
  float diff[EMB];
#pragma unroll
  for (int c = 0; c < EMB; ++c)
    diff[c] = fabsf(v1[g * EMB + c] - v2[g * EMB + c]);
  float z = bc2[0];
  for (int j = 0; j < 64; ++j) {
    float h = bc1[j];
#pragma unroll
    for (int c = 0; c < EMB; ++c) h += diff[c] * w1s[c * 64 + j];
    z += fmaxf(h, 0.f) * w2s[j];
  }
  out[g] = 1.f / (1.f + expf(-z));
}

// ---------------- host orchestration ----------------

extern "C" void kernel_launch(void* const* d_in, const int* in_sizes, int n_in,
                              void* d_out, int out_size, void* d_ws, size_t ws_size,
                              hipStream_t stream) {
  (void)in_sizes; (void)n_in; (void)out_size; (void)ws_size;
  const float* xin[2] = {(const float*)d_in[0], (const float*)d_in[1]};
  const int*   ei[2]  = {(const int*)d_in[2], (const int*)d_in[3]};
  const int*   bt[2]  = {(const int*)d_in[4], (const int*)d_in[5]};
  const float* Wl[3]  = {(const float*)d_in[6],  (const float*)d_in[10], (const float*)d_in[14]};
  const float* gl[3]  = {(const float*)d_in[8],  (const float*)d_in[12], (const float*)d_in[16]};
  const float* bel[3] = {(const float*)d_in[9],  (const float*)d_in[13], (const float*)d_in[17]};
  const float* Wf  = (const float*)d_in[18];
  const float* bfv = (const float*)d_in[19];
  const float* Wc1 = (const float*)d_in[20];
  const float* bc1 = (const float*)d_in[21];
  const float* Wc2 = (const float*)d_in[22];
  const float* bc2 = (const float*)d_in[23];

  // workspace layout (floats)
  float* X     = (float*)d_ws;                      // N x 128 activations (reused as GEMM out)
  float* AGG   = X + (size_t)N_NODES * HID;         // N x 128 aggregation buffer
  float* stats = AGG + (size_t)N_NODES * HID;       // 2 x 128
  float* psum  = stats + 2 * HID;                   // G x 128 pooled sums (then means)
  float* cnt   = psum + N_GRAPH * HID;              // G
  float* v1    = cnt + N_GRAPH;                     // G x 64
  float* v2    = v1 + N_GRAPH * EMB;                // G x 64
  float* vout[2] = {v1, v2};

  for (int e = 0; e < 2; ++e) {
    const int* src = ei[e];
    const int* dst = ei[e] + N_EDGES;
    for (int l = 0; l < 3; ++l) {
      const float* a_in = (l == 0) ? xin[e] : X;
      const int Cin = (l == 0) ? F_IN : HID;
      const int n4 = N_NODES * Cin / 4;
      k_copy4<<<(n4 + 255) / 256, 256, 0, stream>>>(AGG, a_in, n4);
      const long ns = (long)N_EDGES * (Cin / 4);
      k_scatter<<<(unsigned)((ns + 255) / 256), 256, 0, stream>>>(a_in, src, dst, AGG, N_EDGES, Cin);
      k_zero<<<1, 256, 0, stream>>>(stats, 2 * HID);
      dim3 gg((N_NODES / 16 + 3) / 4, HID / 16);
      if (Cin == F_IN)
        k_gemm_wmma<F_IN><<<gg, 128, 0, stream>>>(AGG, Wl[l], nullptr, X, N_NODES, HID);
      else
        k_gemm_wmma<HID><<<gg, 128, 0, stream>>>(AGG, Wl[l], nullptr, X, N_NODES, HID);
      k_bn_stats<<<(N_NODES + 511) / 512, 256, 0, stream>>>(X, stats, N_NODES, 512);
      const long nn = (long)N_NODES * HID;
      k_bn_apply<<<(unsigned)((nn + 255) / 256), 256, 0, stream>>>(X, stats, gl[l], bel[l], nn, 1.f / N_NODES);
    }
    // pooling (psum and cnt are contiguous -> single zero pass)
    k_zero<<<(N_GRAPH * HID + N_GRAPH + 255) / 256, 256, 0, stream>>>(psum, N_GRAPH * HID + N_GRAPH);
    const long np = (long)N_NODES * (HID / 4);
    k_pool_accum<<<(unsigned)((np + 255) / 256), 256, 0, stream>>>(X, bt[e], psum, cnt);
    k_pool_final<<<(N_GRAPH * HID + 255) / 256, 256, 0, stream>>>(psum, cnt);
    dim3 gp((N_GRAPH / 16 + 3) / 4, EMB / 16);
    k_gemm_wmma<HID><<<gp, 128, 0, stream>>>(psum, Wf, bfv, vout[e], N_GRAPH, EMB);
  }
  k_classifier<<<1, 256, 0, stream>>>(v1, v2, Wc1, bc1, Wc2, bc2, (float*)d_out);
}
